// FRNN_27281632264878
// MI455X (gfx1250) — matrix operand order repaired and checked by reading, not compiled, not run on hardware
//
#include <hip/hip_runtime.h>
#include <math.h>

#define T_STEPS   8192
#define VISIBLE   1024
#define FULL      4096
#define NUM_WG    128
#define ROWS_PER_WG 32                      // FULL / NUM_WG
#define THREADS   256                       // 8 waves: 2 M-tiles x 4 K-quarters
#define KT_TOTAL  128                       // FULL / 32 k-tiles
#define MT_PER_WG 2                         // ROWS_PER_WG / 16
#define A_ELEMS   (MT_PER_WG * KT_TOTAL * 32 * 16)   // 131072 bf16 elements
#define A_BYTES   (A_ELEMS * 2)                      // 262144 B of LDS-resident W
#define RED_FLOATS (MT_PER_WG * 4 * 16)              // 128 partial sums
#define SHMEM_BYTES (A_BYTES + RED_FLOATS * 4)

typedef __bf16 v16bf __attribute__((ext_vector_type(16)));
typedef float  v8f   __attribute__((ext_vector_type(8)));

__device__ __forceinline__ unsigned short f32_to_bf16(float f) {
  unsigned int u = __float_as_uint(f);
  u += 0x7FFFu + ((u >> 16) & 1u);          // round to nearest even
  return (unsigned short)(u >> 16);
}

extern "C" __global__ __launch_bounds__(THREADS, 1)
void frnn_persistent(const float* __restrict__ X,
                     const float* __restrict__ W,
                     const float* __restrict__ Lam,
                     float* __restrict__ out,
                     unsigned int* __restrict__ cnt,     // [T_STEPS] step counters
                     unsigned short* __restrict__ rbuf)  // [2][FULL] bf16 r, double buffered
{
  extern __shared__ __align__(32) unsigned char smem_raw[];
  unsigned short* ldsA = (unsigned short*)smem_raw;     // W slice, WMMA-A fragment order
  float* red = (float*)(smem_raw + A_BYTES);            // cross-wave reduction buffer

  const int tid    = threadIdx.x;
  const int lane   = tid & 31;
  const int wave   = tid >> 5;
  const int laneHi = lane >> 4;
  const int rowBase = blockIdx.x * ROWS_PER_WG;

  // ---- one-time staging: fp32 W slice -> bf16 LDS, pre-swizzled to a
  //      16x32 A-fragment layout, contiguous k = 16*half + j ------------------------------
  for (int e = tid; e < A_ELEMS; e += THREADS) {
    int j  = e & 15;                // element within lane's 16-bf16 fragment
    int ln = (e >> 4) & 31;         // lane
    int q  = e >> 9;
    int kt = q & (KT_TOTAL - 1);    // k-tile
    int mt = q >> 7;                // m-tile
    int lh = ln >> 4;
    int M  = ln & 15;
    int K  = j + 16 * lh;                       // contiguous K mapping (k = 16*half + j), matches B
    ldsA[e] = f32_to_bf16(W[(size_t)(rowBase + mt * 16 + M) * FULL + kt * 32 + K]);
  }
  __syncthreads();

  const int mt     = wave >> 2;     // this wave's 16-row tile (0/1)
  const int kq     = wave & 3;      // this wave's K quarter
  const int ktBase = kq * 32;       // 32 k-tiles of 32 => K range 1024

  for (int t = 0; t < T_STEPS; ++t) {
    // wait until r_t (produced by step t-1, all 128 WGs) is globally visible
    if (t > 0) {
      if (tid == 0) {
        while (__hip_atomic_load(&cnt[t - 1], __ATOMIC_ACQUIRE,
                                 __HIP_MEMORY_SCOPE_AGENT) < NUM_WG) {
          __builtin_amdgcn_s_sleep(2);
        }
      }
      __syncthreads();
    }

    const unsigned short* __restrict__ rprev = rbuf + (size_t)(t & 1) * FULL;
    unsigned short* __restrict__ rnext = rbuf + (size_t)((t + 1) & 1) * FULL;

    // ---- 32 WMMAs per wave: u_tile += W_tile (LDS) x broadcast(r) --------------
    v8f c0 = {};
    v8f c1 = {};
    const unsigned short* aTile =
        ldsA + ((size_t)(mt * KT_TOTAL + ktBase) * 32 + lane) * 16;
    const unsigned short* bPtr = rprev + ktBase * 32 + laneHi * 16;
#pragma unroll
    for (int i = 0; i < 32; i += 2) {
      v16bf a0 = *(const v16bf*)(aTile + (size_t)i * (32 * 16));
      v16bf b0 = *(const v16bf*)(bPtr + (size_t)i * 32);
      c0 = __builtin_amdgcn_wmma_f32_16x16x32_bf16(false, a0, false, b0,
                                                   (short)0, c0, false, false);
      v16bf a1 = *(const v16bf*)(aTile + (size_t)(i + 1) * (32 * 16));
      v16bf b1 = *(const v16bf*)(bPtr + (size_t)(i + 1) * 32);
      c1 = __builtin_amdgcn_wmma_f32_16x16x32_bf16(false, a1, false, b1,
                                                   (short)0, c1, false, false);
    }
    v8f acc = c0 + c1;

    // ---- cross-wave reduction: all N columns identical, take column 0 ----------
    // D layout: lanes 0-15 vgpr g -> row g, lanes 16-31 vgpr g -> row 8+g
    if (lane == 0) {
#pragma unroll
      for (int g = 0; g < 8; ++g) red[(mt * 4 + kq) * 16 + g] = acc[g];
    } else if (lane == 16) {
#pragma unroll
      for (int g = 0; g < 8; ++g) red[(mt * 4 + kq) * 16 + 8 + g] = acc[g];
    }
    __syncthreads();

    // ---- gate + tanh + publish r chunk ----------------------------------------
    if (tid < ROWS_PER_WG) {
      int row = tid;
      int m = rowBase + row;
      const float* rb = red + (row >> 4) * 64 + (row & 15);
      float s = rb[0] + rb[16] + rb[32] + rb[48];
      float lam = Lam[m];
      float xin = 0.0f;
      if (m < VISIBLE) {
        xin = X[(size_t)t * VISIBLE + m];
        if (t + 1 < T_STEPS)
          __builtin_prefetch(&X[(size_t)(t + 1) * VISIBLE + m], 0, 0);
      }
      float u = lam * s + (1.0f - lam) * xin;
      rnext[m] = f32_to_bf16(tanhf(u));
      if (t == T_STEPS - 1 && m < VISIBLE) out[m] = u;
    }
    __threadfence();          // make r chunk device-visible before signaling
    __syncthreads();
    if (tid == 0)
      __hip_atomic_fetch_add(&cnt[t], 1u, __ATOMIC_RELEASE,
                             __HIP_MEMORY_SCOPE_AGENT);
  }
}

extern "C" void kernel_launch(void* const* d_in, const int* in_sizes, int n_in,
                              void* d_out, int out_size, void* d_ws, size_t ws_size,
                              hipStream_t stream) {
  const float* X   = (const float*)d_in[0];   // [8192, 1024]
  const float* W   = (const float*)d_in[1];   // [4096, 4096]
  const float* Lam = (const float*)d_in[2];   // [4096]
  float* out = (float*)d_out;                 // [1024]

  unsigned int*   cnt  = (unsigned int*)d_ws;                        // 32 KB
  unsigned short* rbuf = (unsigned short*)((char*)d_ws + 32768);     // 16 KB

  // zero step counters and r0 (= tanh(0)) each call: graph-replay deterministic
  hipMemsetAsync(d_ws, 0, 32768 + 16384, stream);

  // 256 KB LDS-resident weight slice per WG (WGP has 320 KB)
  hipFuncSetAttribute((const void*)frnn_persistent,
                      hipFuncAttributeMaxDynamicSharedMemorySize, SHMEM_BYTES);

  hipLaunchKernelGGL(frnn_persistent, dim3(NUM_WG), dim3(THREADS), SHMEM_BYTES,
                     stream, X, W, Lam, out, cnt, rbuf);
}
